// KanervaMemory_14757507629483
// MI455X (gfx1250) — compile-verified, matrix operand order checked by reading, not run on hardware
//
#include <hip/hip_runtime.h>
#include <hip/hip_bf16.h>
#include <math.h>

// ---------------------------------------------------------------------------
// KanervaMemory pipeline for MI455X (gfx1250, wave32, WMMA bf16 16x16x32).
// Dominant cost: 16896^2 x 256 distance GEMM (~146 GFLOP) -> bf16 WMMA,
// candidate set (8.6MB bf16) resident in 192MB L2.  All other GEMMs reuse the
// same X @ W^T WMMA tile primitive.  deg==9 identity collapses the GCN step.
// ---------------------------------------------------------------------------

#define Z     256
#define RLEN  512
#define MEML  16384
#define NTOT  16896   // RLEN + MEML
#define NQ    1024
#define KNN   7
#define NH    8
#define DH    32      // Z / NH
#define SG    7168    // NQ * KNN

typedef __attribute__((ext_vector_type(16))) __bf16 v16bf;
typedef __attribute__((ext_vector_type(8)))  float  v8f;

// ---- WMMA helpers ---------------------------------------------------------

__device__ __forceinline__ v8f wmma_bf16(v16bf a, v16bf b, v8f c) {
  // D = A(16x32 bf16) x B(32x16 bf16) + C(16x16 f32)
  return __builtin_amdgcn_wmma_f32_16x16x32_bf16(false, a, false, b,
                                                 (short)0, c, false, false);
}

// A-fragment (16x32, bf16) from a row-major f32 matrix (converted on the fly).
// Lane L (L<16): row M=L, K = {k0..k0+7, k0+16..k0+23}; L>=16: row M=L-16,
// K = {k0+8..k0+15, k0+24..k0+31}.  B for X@W^T uses the identical pattern.
__device__ __forceinline__ v16bf frag_f32(const float* X, int ld, int row0, int k0) {
  const int lane = threadIdx.x & 31;
  const float* p = X + (size_t)(row0 + (lane & 15)) * ld + k0 + ((lane >> 4) ? 8 : 0);
  v16bf a;
#pragma unroll
  for (int i = 0; i < 8; ++i) a[i] = (__bf16)p[i];
#pragma unroll
  for (int i = 0; i < 8; ++i) a[8 + i] = (__bf16)p[16 + i];
  return a;
}

__device__ __forceinline__ v16bf frag_bf16(const __bf16* X, int ld, int row0, int k0) {
  const int lane = threadIdx.x & 31;
  const __bf16* p = X + (size_t)(row0 + (lane & 15)) * ld + k0 + ((lane >> 4) ? 8 : 0);
  v16bf a;
#pragma unroll
  for (int i = 0; i < 8; ++i) a[i] = p[i];
#pragma unroll
  for (int i = 0; i < 8; ++i) a[8 + i] = p[16 + i];
  return a;
}

// ---- Generic GEMM: C[M x N] = X[M x K] @ W[N x K]^T  (optional row gather) -
__global__ void gemm_xwT_kernel(const float* __restrict__ X,
                                const float* __restrict__ W,
                                float* __restrict__ C,
                                const int* __restrict__ rowidx,
                                int M, int N, int Kd) {
  const int wave = (blockIdx.x * (blockDim.x >> 5)) + (threadIdx.x >> 5);
  const int ntn = N >> 4;
  const int total = (M >> 4) * ntn;
  if (wave >= total) return;
  const int m0 = (wave / ntn) << 4;
  const int n0 = (wave % ntn) << 4;
  const int lane = threadIdx.x & 31;

  int arow = m0 + (lane & 15);
  if (rowidx) arow = rowidx[arow];
  const float* pa = X + (size_t)arow * Kd + ((lane >> 4) ? 8 : 0);
  const float* pb = W + (size_t)(n0 + (lane & 15)) * Kd + ((lane >> 4) ? 8 : 0);

  v8f acc = {};
  for (int k0 = 0; k0 < Kd; k0 += 32) {
    v16bf a, b;
#pragma unroll
    for (int i = 0; i < 8; ++i) { a[i] = (__bf16)pa[k0 + i];      b[i] = (__bf16)pb[k0 + i]; }
#pragma unroll
    for (int i = 0; i < 8; ++i) { a[8 + i] = (__bf16)pa[k0 + 16 + i]; b[8 + i] = (__bf16)pb[k0 + 16 + i]; }
    acc = wmma_bf16(a, b, acc);
  }
  const int hi = (lane >> 4);
#pragma unroll
  for (int v = 0; v < 8; ++v)
    C[(size_t)(m0 + v + hi * 8) * N + n0 + (lane & 15)] = acc[v];
}

// ---- q_y_sample: out = mu + (1e-6 + softplus(s)) * eps --------------------
// mu = X@W[0:Z]^T + b[0:Z], s = X@W[Z:2Z]^T + b[Z:2Z]
__global__ void qy_kernel(const float* __restrict__ X,
                          const float* __restrict__ eps,
                          const float* __restrict__ W,
                          const float* __restrict__ b,
                          float* __restrict__ out, int M) {
  const int wave = (blockIdx.x * (blockDim.x >> 5)) + (threadIdx.x >> 5);
  const int ntn = Z >> 4;
  const int total = (M >> 4) * ntn;
  if (wave >= total) return;
  const int m0 = (wave / ntn) << 4;
  const int c0 = (wave % ntn) << 4;
  const int lane = threadIdx.x & 31;

  v8f accm = {}, accs = {};
  for (int k0 = 0; k0 < Z; k0 += 32) {
    v16bf a  = frag_f32(X, Z, m0, k0);
    v16bf bm = frag_f32(W, Z, c0, k0);
    v16bf bs = frag_f32(W, Z, Z + c0, k0);
    accm = wmma_bf16(a, bm, accm);
    accs = wmma_bf16(a, bs, accs);
  }
  const int hi = (lane >> 4);
  const int col = c0 + (lane & 15);
#pragma unroll
  for (int v = 0; v < 8; ++v) {
    const int row = m0 + v + hi * 8;
    float mu = accm[v] + b[col];
    float s  = accs[v] + b[Z + col];
    float sp = (s > 20.f) ? s : log1pf(__expf(s));
    out[(size_t)row * Z + col] = mu + (1e-6f + sp) * eps[(size_t)row * Z + col];
  }
}

// ---- row prep: bf16 copy + squared norms ----------------------------------
__global__ void rowprep_kernel(const float* __restrict__ X,
                               __bf16* __restrict__ Xb,
                               float* __restrict__ x2, int M) {
  const int row = blockIdx.x * (blockDim.x >> 5) + (threadIdx.x >> 5);
  if (row >= M) return;
  const int lane = threadIdx.x & 31;
  float s = 0.f;
#pragma unroll
  for (int i = lane; i < Z; i += 32) {
    float v = X[(size_t)row * Z + i];
    Xb[(size_t)row * Z + i] = (__bf16)v;
    s += v * v;
  }
#pragma unroll
  for (int m = 1; m < 32; m <<= 1) s += __shfl_xor(s, m, 32);
  if (lane == 0) x2[row] = s;
}

// ---- KNN (top-7 smallest of p2[j] - 2*q.p) --------------------------------
// One wave per 16-query tile.  A-frags for all of K=256 held in registers.
__global__ void knn_kernel(const __bf16* __restrict__ Qb,
                           const __bf16* __restrict__ Pb,
                           const float* __restrict__ p2,
                           int Np, int excludeSelf,
                           int* __restrict__ outIdx) {
  __shared__ float lds_d[16][17];
  __shared__ float md[16][16];
  __shared__ int   mi[16][16];

  const int lane = threadIdx.x & 31;
  const int m0 = blockIdx.x << 4;
  const int hi = (lane >> 4);
  const int rowl = lane & 15;

  v16bf afr[8];
#pragma unroll
  for (int kk = 0; kk < 8; ++kk) afr[kk] = frag_bf16(Qb, Z, m0, kk * 32);

  float bestd[KNN];
  int   besti[KNN];
#pragma unroll
  for (int t = 0; t < KNN; ++t) { bestd[t] = 3.4e38f; besti[t] = 0; }

  const int mg_scan = m0 + rowl;          // query row this lane scans for
  const int cbase = hi ? 8 : 0;

  for (int n0 = 0; n0 < Np; n0 += 16) {
    if (n0 + 16 < Np)
      __builtin_prefetch(Pb + (size_t)(n0 + 16) * Z + (size_t)lane * 16, 0, 1);

    v8f acc = {};
#pragma unroll
    for (int kk = 0; kk < 8; ++kk)
      acc = wmma_bf16(afr[kk], frag_bf16(Pb, Z, n0, kk * 32), acc);

    const float pj = p2[n0 + rowl];       // this lane's candidate column norm
#pragma unroll
    for (int v = 0; v < 8; ++v)
      lds_d[v + hi * 8][rowl] = pj - 2.f * acc[v];
    __syncthreads();

#pragma unroll
    for (int c = 0; c < 8; ++c) {
      const int j = n0 + cbase + c;
      float d = lds_d[rowl][cbase + c];
      if (excludeSelf && j == mg_scan) d = 3.4e38f;
      if (d < bestd[KNN - 1]) {
        bestd[KNN - 1] = d; besti[KNN - 1] = j;
#pragma unroll
        for (int p = KNN - 1; p > 0; --p) {
          if (bestd[p] < bestd[p - 1]) {
            float td = bestd[p]; bestd[p] = bestd[p - 1]; bestd[p - 1] = td;
            int ti = besti[p];   besti[p] = besti[p - 1]; besti[p - 1] = ti;
          }
        }
      }
    }
    __syncthreads();
  }

  // merge the two sorted half-lists (cols 0..7 owner and 8..15 owner)
#pragma unroll
  for (int t = 0; t < KNN; ++t) { md[rowl][hi * KNN + t] = bestd[t]; mi[rowl][hi * KNN + t] = besti[t]; }
  __syncthreads();
  if (lane < 16) {
    int ia = 0, ib = KNN;
    for (int t = 0; t < KNN; ++t) {
      float da = (ia < KNN)     ? md[lane][ia] : 3.4e38f;
      float db = (ib < 2 * KNN) ? md[lane][ib] : 3.4e38f;
      int pick;
      if (da <= db) { pick = mi[lane][ia]; ++ia; }
      else          { pick = mi[lane][ib]; ++ib; }
      outIdx[(size_t)(m0 + lane) * KNN + t] = pick;
    }
  }
}

// ---- GCN gather: mem_vals[i] = (2*xw[i+RLEN] + sum_nbr xw[j]) / 9 ---------
__global__ void gcn_gather_kernel(const float* __restrict__ xw,
                                  const int* __restrict__ nbr,
                                  float* __restrict__ mem_vals) {
  const size_t total = (size_t)MEML * Z;
  for (size_t t = (size_t)blockIdx.x * blockDim.x + threadIdx.x; t < total;
       t += (size_t)gridDim.x * blockDim.x) {
    const int i = (int)(t / Z);
    const int c = (int)(t % Z);
    const int node = i + RLEN;
    float s = 2.f * xw[(size_t)node * Z + c];
#pragma unroll
    for (int k = 0; k < KNN; ++k)
      s += xw[(size_t)nbr[(size_t)node * KNN + k] * Z + c];
    mem_vals[(size_t)i * Z + c] = s * (1.f / 9.f);
  }
}

// ---- attention: per (head, 16-query tile), two-pass online softmax --------
__global__ void attn_kernel(const float* __restrict__ Q,
                            const float* __restrict__ Kp,
                            const float* __restrict__ Vp,
                            float* __restrict__ O) {
  __shared__ __bf16 plds[16][40];

  const int lane = threadIdx.x & 31;
  const int h = blockIdx.y;
  const int l0 = blockIdx.x << 4;
  const int hi = (lane >> 4);
  const int rowl = lane & 15;
  const float scale = rsqrtf((float)DH);

  const v16bf qf = frag_f32(Q, Z, l0, h * DH);   // K = dh = 32: single frag

  // pass 1: row maxima
  float rm[8];
#pragma unroll
  for (int v = 0; v < 8; ++v) rm[v] = -3.4e38f;
  for (int s0 = 0; s0 < SG; s0 += 16) {
    v8f acc = {};
    acc = wmma_bf16(qf, frag_f32(Kp, Z, s0, h * DH), acc);
#pragma unroll
    for (int v = 0; v < 8; ++v) rm[v] = fmaxf(rm[v], acc[v] * scale);
  }
#pragma unroll
  for (int v = 0; v < 8; ++v)
#pragma unroll
    for (int m = 1; m < 16; m <<= 1) rm[v] = fmaxf(rm[v], __shfl_xor(rm[v], m, 32));

  // pass 2: sumexp + P@V
  float rl[8] = {0.f, 0.f, 0.f, 0.f, 0.f, 0.f, 0.f, 0.f};
  v8f o0 = {}, o1 = {};
  const int kb = hi ? 8 : 0;
  const int ccol = h * DH + rowl;

  for (int s0 = 0; s0 < SG; s0 += 32) {
#pragma unroll
    for (int t = 0; t < 2; ++t) {
      v8f acc = {};
      acc = wmma_bf16(qf, frag_f32(Kp, Z, s0 + t * 16, h * DH), acc);
#pragma unroll
      for (int v = 0; v < 8; ++v) {
        float p = __expf(acc[v] * scale - rm[v]);
        rl[v] += p;
        plds[v + hi * 8][t * 16 + rowl] = (__bf16)p;
      }
    }
    __syncthreads();

    v16bf pf, vf0, vf1;
#pragma unroll
    for (int i = 0; i < 8; ++i) {
      pf[i]     = plds[rowl][kb + i];
      pf[8 + i] = plds[rowl][kb + 16 + i];
      vf0[i]     = (__bf16)Vp[(size_t)(s0 + kb + i) * Z + ccol];
      vf0[8 + i] = (__bf16)Vp[(size_t)(s0 + kb + 16 + i) * Z + ccol];
      vf1[i]     = (__bf16)Vp[(size_t)(s0 + kb + i) * Z + ccol + 16];
      vf1[8 + i] = (__bf16)Vp[(size_t)(s0 + kb + 16 + i) * Z + ccol + 16];
    }
    o0 = wmma_bf16(pf, vf0, o0);
    o1 = wmma_bf16(pf, vf1, o1);
    __syncthreads();
  }

#pragma unroll
  for (int v = 0; v < 8; ++v)
#pragma unroll
    for (int m = 1; m < 16; m <<= 1) rl[v] += __shfl_xor(rl[v], m, 32);

#pragma unroll
  for (int v = 0; v < 8; ++v) {
    const int row = l0 + v + hi * 8;
    const float inv = 1.f / rl[v];
    O[(size_t)row * Z + h * DH + rowl]      = o0[v] * inv;
    O[(size_t)row * Z + h * DH + 16 + rowl] = o1[v] * inv;
  }
}

// ---------------------------------------------------------------------------
extern "C" void kernel_launch(void* const* d_in, const int* in_sizes, int n_in,
                              void* d_out, int out_size, void* d_ws, size_t ws_size,
                              hipStream_t stream) {
  const float* r         = (const float*)d_in[0];
  const float* y         = (const float*)d_in[1];
  const float* eps_write = (const float*)d_in[2];
  const float* eps_keys  = (const float*)d_in[3];
  const float* keys      = (const float*)d_in[4];
  const float* vals      = (const float*)d_in[5];
  const float* W_ry      = (const float*)d_in[6];
  const float* b_ry      = (const float*)d_in[7];
  const float* W_gcn     = (const float*)d_in[8];
  const float* W_in      = (const float*)d_in[9];
  const float* W_out     = (const float*)d_in[10];
  float* out = (float*)d_out;

  char* ws = (char*)d_ws;
  size_t o = 0;
  auto alloc = [&](size_t bytes) {
    size_t cur = o;
    o = (o + bytes + 255) & ~(size_t)255;
    return cur;
  };
  float*  new_keys = (float*)(ws + alloc((size_t)NTOT * Z * 4));
  float*  new_vals = (float*)(ws + alloc((size_t)NTOT * Z * 4));
  __bf16* nkb      = (__bf16*)(ws + alloc((size_t)NTOT * Z * 2));
  float*  nk2      = (float*)(ws + alloc((size_t)NTOT * 4));
  int*    nbr      = (int*)  (ws + alloc((size_t)NTOT * KNN * 4));
  float*  xw       = (float*)(ws + alloc((size_t)NTOT * Z * 4));
  float*  mem_vals = (float*)(ws + alloc((size_t)MEML * Z * 4));
  float*  mem_keys = (float*)(ws + alloc((size_t)MEML * Z * 4));
  __bf16* mkb      = (__bf16*)(ws + alloc((size_t)MEML * Z * 2));
  float*  mk2      = (float*)(ws + alloc((size_t)MEML * 4));
  __bf16* yb       = (__bf16*)(ws + alloc((size_t)NQ * Z * 2));
  float*  y2       = (float*)(ws + alloc((size_t)NQ * 4));
  int*    idxb     = (int*)  (ws + alloc((size_t)NQ * KNN * 4));
  float*  Qb       = (float*)(ws + alloc((size_t)NQ * Z * 4));
  float*  Kb       = (float*)(ws + alloc((size_t)SG * Z * 4));
  float*  Vb       = (float*)(ws + alloc((size_t)SG * Z * 4));
  float*  attout   = (float*)(ws + alloc((size_t)NQ * Z * 4));

  // 1) concatenations (device-to-device async copies are graph-capturable)
  hipMemcpyAsync(new_vals,            r,    (size_t)RLEN * Z * 4, hipMemcpyDeviceToDevice, stream);
  hipMemcpyAsync(new_vals + (size_t)RLEN * Z, vals, (size_t)MEML * Z * 4, hipMemcpyDeviceToDevice, stream);
  hipMemcpyAsync(new_keys + (size_t)RLEN * Z, keys, (size_t)MEML * Z * 4, hipMemcpyDeviceToDevice, stream);

  // 2) yw = q_y_sample(r) -> new_keys[0:512]
  qy_kernel<<<(RLEN / 16) * (Z / 16) / 4, 128, 0, stream>>>(r, eps_write, W_ry, b_ry, new_keys, RLEN);

  // 3) bf16 + norms for the graph candidates
  rowprep_kernel<<<NTOT / 8, 256, 0, stream>>>(new_keys, nkb, nk2, NTOT);

  // 4) KNN graph over all 16896 points (dominant: 146-GFLOP bf16 WMMA GEMM)
  knn_kernel<<<NTOT / 16, 32, 0, stream>>>(nkb, nkb, nk2, NTOT, 1, nbr);

  // 5) xw = new_vals @ W_gcn^T
  gemm_xwT_kernel<<<(NTOT / 16) * (Z / 16) / 4, 128, 0, stream>>>(new_vals, W_gcn, xw, nullptr, NTOT, Z, Z);

  // 6) GCN aggregate (deg==9 identity), only mem rows
  gcn_gather_kernel<<<4096, 256, 0, stream>>>(xw, nbr, mem_vals);

  // 7) mem_keys = q_y_sample(mem_vals)
  qy_kernel<<<(MEML / 16) * (Z / 16) / 4, 128, 0, stream>>>(mem_vals, eps_keys, W_ry, b_ry, mem_keys, MEML);

  // 8) prep for query KNN
  rowprep_kernel<<<MEML / 8, 256, 0, stream>>>(mem_keys, mkb, mk2, MEML);
  rowprep_kernel<<<NQ / 8,   256, 0, stream>>>(y, yb, y2, NQ);

  // 9) idx = knn(mem_keys, y, 7)
  knn_kernel<<<NQ / 16, 32, 0, stream>>>(yb, mkb, mk2, MEML, 0, idxb);

  // 10) Q/K/V projections (K/V fused with the idx gather)
  gemm_xwT_kernel<<<(NQ / 16) * (Z / 16) / 4, 128, 0, stream>>>(y, W_in, Qb, nullptr, NQ, Z, Z);
  gemm_xwT_kernel<<<(SG / 16) * (Z / 16) / 4, 128, 0, stream>>>(mem_keys, W_in + (size_t)Z * Z,     Kb, idxb, SG, Z, Z);
  gemm_xwT_kernel<<<(SG / 16) * (Z / 16) / 4, 128, 0, stream>>>(mem_vals, W_in + (size_t)2 * Z * Z, Vb, idxb, SG, Z, Z);

  // 11) attention (8 heads x 64 query tiles)
  attn_kernel<<<dim3(NQ / 16, NH), 32, 0, stream>>>(Qb, Kb, Vb, attout);

  // 12) out = attout @ W_out^T
  gemm_xwT_kernel<<<(NQ / 16) * (Z / 16) / 4, 128, 0, stream>>>(attout, W_out, out, nullptr, NQ, Z, Z);

  (void)in_sizes; (void)n_in; (void)out_size; (void)ws_size;
}
